// SingleMultiViewModel_81097572483185
// MI455X (gfx1250) — compile-verified, matrix operand order checked by reading, not run on hardware
//
#include <hip/hip_runtime.h>
#include <hip/hip_bf16.h>

typedef __attribute__((ext_vector_type(16))) __bf16 v16bf;
typedef __attribute__((ext_vector_type(8)))  __bf16 v8bf;
typedef __attribute__((ext_vector_type(2)))  __bf16 v2bf;
typedef __attribute__((ext_vector_type(8)))  float  v8f;
typedef __attribute__((ext_vector_type(4)))  float  v4f;

#define WAVES 8
#define BLK   256

// ---------------------------------------------------------------------------
// Fragment index helpers.
// A-fragment (16x32 bf16): lane = hl*16 + (M&15); element j -> K:
//   K32 = (j<8) ? hl*8 + j : 16 + hl*8 + (j-8)
// B-fragment (32x16 bf16): mirrored with N in place of M.
// C/D (16x16 f32, 8 VGPRs): lane -> N=lane&15, hl=lane>>4; vgpr r -> M=hl*8+r.
// Key epilogue fact: for fixed (mt,lane), the 8 C rows map to the SAME target
// lane and 8 consecutive bf16 elements (j = r + 8*(mt&1)) -> one b128 store.
// ---------------------------------------------------------------------------
__device__ __host__ __forceinline__ int kmap(int hl, int j) {
    return (j < 8) ? (hl * 8 + j) : (16 + hl * 8 + (j - 8));
}

// ---------------------------------------------------------------------------
// Kernel A: permute an [M x K] fp32 row-major weight into bf16 A-fragments
// laid out [mt][kt][lane][16] so GEMM loads are contiguous 32B per lane.
// ---------------------------------------------------------------------------
__global__ void permute_weight(const float* __restrict__ W, __bf16* __restrict__ dst,
                               int M, int K, int Mt, int Kt) {
    int e = blockIdx.x * BLK + threadIdx.x;
    int total = Mt * Kt * 512;
    if (e >= total) return;
    int frag = e >> 9, idx = e & 511;
    int lane = idx >> 4, j = idx & 15;
    int mt = frag / Kt, kt = frag % Kt;
    int hl = lane >> 4;
    int Mg = mt * 16 + (lane & 15);
    int Kg = kt * 32 + kmap(hl, j);
    float v = (Mg < M && Kg < K) ? W[(size_t)Mg * K + Kg] : 0.0f;
    dst[e] = (__bf16)v;
}

// ---------------------------------------------------------------------------
// Fused per-batch-element pipeline
// ---------------------------------------------------------------------------
struct Params {
    const float* x;        // [4096,120,5,1,5]
    float*       out;      // same shape
    const __bf16 *wExp, *wL0, *wL1, *wPs, *wPsS, *wAs, *wAsS;  // permuted
    const float *bnE_g, *bnE_b, *bnE_m, *bnE_v;
    const float *bn0_g, *bn0_b, *bn0_m, *bn0_v;
    const float *bn1_g, *bn1_b, *bn1_m, *bn1_v;
    const float *bnP_g, *bnP_b, *bnP_m, *bnP_v;
    const float *bnA_g, *bnA_b, *bnA_m, *bnA_v;
    const float *psb;      // [25]
    const float *asb;      // [600]
};

// Folds BN to (s,t); writes s=t=0 for padding rows C..303 so the GEMM epilogue
// produces exact zeros there (keeps next layer's K padding clean, no guards).
__device__ __forceinline__ void stage_bn(const float* g, const float* b,
                                         const float* m, const float* v,
                                         float* sb, int tid, int C) {
    for (int i = tid; i < 304; i += BLK) {
        float s = 0.0f, t = 0.0f;
        if (i < C) {
            s = g[i] * rsqrtf(v[i] + 1e-5f);
            t = b[i] - m[i] * s;
        }
        sb[i * 2]     = s;
        sb[i * 2 + 1] = t;
    }
}

// MODE 0: BN+ReLU -> bf16 B-fragments (packed b128 stores, no row guards).
// MODE 1: + residual add (packed b128 res loads).
// MODE 2: +bias -> plain f32 [row][32] (cols < 25 only).
template <int MODE>
__device__ __forceinline__ void gemm_stage(
    const __bf16* __restrict__ Wg, int Mt, int Kt, int Mrows,
    const __bf16* inFrag, __bf16* outFrag, const __bf16* resFrag,
    float* outPlain, const float* sb, const float* __restrict__ biasG,
    int lane, int wave) {
    int n = lane & 15, hl = lane >> 4;
    for (int mt = wave; mt < Mt; mt += WAVES) {
        v8f c0 = {}; v8f c1 = {};
        const __bf16* wrow = Wg + (size_t)mt * Kt * 512 + lane * 16;
        // hint next M-tile's weights toward the caches
        __builtin_prefetch(wrow + (size_t)WAVES * Kt * 512, 0, 1);
        for (int kt = 0; kt < Kt; ++kt) {
            v16bf a  = *(const v16bf*)(wrow + (size_t)kt * 512);
            v16bf b0 = *(const v16bf*)(inFrag + (kt * 2 + 0) * 512 + lane * 16);
            v16bf b1 = *(const v16bf*)(inFrag + (kt * 2 + 1) * 512 + lane * 16);
            c0 = __builtin_amdgcn_wmma_f32_16x16x32_bf16(false, a, false, b0,
                                                         (short)0, c0, false, false);
            c1 = __builtin_amdgcn_wmma_f32_16x16x32_bf16(false, a, false, b1,
                                                         (short)0, c1, false, false);
        }
        int rowBase = mt * 16 + hl * 8;
        if (MODE == 2) {
#pragma unroll
            for (int r = 0; r < 8; ++r) {
                int row = rowBase + r;
                if (row >= Mrows) continue;
                float bs = biasG[row];
                if (n < 25)      outPlain[row * 32 + n]      = c0[r] + bs;
                if (16 + n < 25) outPlain[row * 32 + 16 + n] = c1[r] + bs;
            }
        } else {
            // packed epilogue: 8 rows -> 8 consecutive bf16 in target fragment
            int kt2 = rowBase >> 5;
            int jb  = (rowBase & 16) ? 8 : 0;
            int o0  = (kt2 * 2 + 0) * 512 + (hl * 16 + n) * 16 + jb;
            int o1  = (kt2 * 2 + 1) * 512 + (hl * 16 + n) * 16 + jb;
            // 16 consecutive (s,t) floats, always in-bounds (sb sized 304*2)
            float sbv[16];
            *(v4f*)&sbv[0]  = *(const v4f*)(sb + rowBase * 2 + 0);
            *(v4f*)&sbv[4]  = *(const v4f*)(sb + rowBase * 2 + 4);
            *(v4f*)&sbv[8]  = *(const v4f*)(sb + rowBase * 2 + 8);
            *(v4f*)&sbv[12] = *(const v4f*)(sb + rowBase * 2 + 12);
            v8bf p0, p1;
#pragma unroll
            for (int r = 0; r < 8; ++r) {
                float s = sbv[2 * r], t = sbv[2 * r + 1];
                p0[r] = (__bf16)fmaxf(fmaf(c0[r], s, t), 0.0f);
                p1[r] = (__bf16)fmaxf(fmaf(c1[r], s, t), 0.0f);
            }
            if (MODE == 1) {
                v8bf r0 = *(const v8bf*)(resFrag + o0);
                v8bf r1 = *(const v8bf*)(resFrag + o1);
#pragma unroll
                for (int r = 0; r < 8; ++r) {
                    p0[r] = (__bf16)((float)p0[r] + (float)r0[r]);
                    p1[r] = (__bf16)((float)p1[r] + (float)r1[r]);
                }
            }
            *(v8bf*)(outFrag + o0) = p0;
            *(v8bf*)(outFrag + o1) = p1;
        }
    }
}

__global__ __launch_bounds__(BLK) void fused_multiview(Params P) {
    // ---- LDS (~154 KB) ----
    __shared__ __align__(32) __bf16 s_xp[38 * 2 * 512];        // 77,824 B (xp; later att f32)
    __shared__ __align__(32) __bf16 s_act[3][10 * 2 * 512];    // 61,440 B
    __shared__ __align__(16) float s_x[3000];                   // 12,000 B
    __shared__ __align__(16) float s_sb[304 * 2];               //  2,432 B
    __shared__ float s_p[25 * 32];                              //  3,200 B
    float* s_att = (float*)s_xp;                                // [600][32]

    const int b    = blockIdx.x;
    const int tid  = threadIdx.x;
    const int lane = tid & 31;
    const int wave = tid >> 5;

    // 0) zero bf16 act buffers (K padding rows 304..319 must be 0, not poison)
    {
        v8bf z = {};
        for (int i = tid; i < 3 * 10 * 2 * 64; i += BLK)
            ((v8bf*)s_act)[i] = z;
    }

    // 1) stage x[b]  (flat [ch=c1*5+c2][view], 600x5) with 16B vector loads
    {
        const v4f* src = (const v4f*)(P.x + (size_t)b * 3000);
        for (int i = tid; i < 750; i += BLK) ((v4f*)s_x)[i] = src[i];
    }

    stage_bn(P.bnE_g, P.bnE_b, P.bnE_m, P.bnE_v, s_sb, tid, 300);
    __syncthreads();

    // 2) build xp B-fragments: 38 k-tiles x 2 n-tiles, bf16 pairs (b32 stores)
    for (int e2 = tid; e2 < 38 * 2 * 256; e2 += BLK) {
        int e = e2 << 1;
        int frag = e >> 9, idx = e & 511;
        int ln = idx >> 4, j = idx & 15;     // j even -> K0,K0+1 adjacent
        int kt = frag >> 1, nt = frag & 1;
        int hl = ln >> 4;
        int col = nt * 16 + (ln & 15);       // pair index p = s*5 + m
        float v0 = 0.0f, v1 = 0.0f;
        if (col < 25) {
            int s = col / 5, m = col % 5;
            int K0 = kt * 32 + kmap(hl, j);
            int K1 = K0 + 1;
            if (K0 < 1200) {
                int view = (K0 < 600) ? s : m;
                int ch   = (K0 < 600) ? K0 : K0 - 600;
                v0 = s_x[ch * 5 + view];
            }
            if (K1 < 1200) {
                int view = (K1 < 600) ? s : m;
                int ch   = (K1 < 600) ? K1 : K1 - 600;
                v1 = s_x[ch * 5 + view];
            }
        }
        v2bf p; p[0] = (__bf16)v0; p[1] = (__bf16)v1;
        *(v2bf*)(s_xp + e) = p;
    }
    __syncthreads();

    // 3) expand [300x1200] -> actA (res)
    gemm_stage<0>(P.wExp, 19, 38, 300, s_xp, s_act[0], nullptr, nullptr, s_sb,
                  nullptr, lane, wave);
    __syncthreads();
    stage_bn(P.bn0_g, P.bn0_b, P.bn0_m, P.bn0_v, s_sb, tid, 300);
    __syncthreads();

    // 4) l0 [300x300]: actA -> actB
    gemm_stage<0>(P.wL0, 19, 10, 300, s_act[0], s_act[1], nullptr, nullptr, s_sb,
                  nullptr, lane, wave);
    __syncthreads();
    stage_bn(P.bn1_g, P.bn1_b, P.bn1_m, P.bn1_v, s_sb, tid, 300);
    __syncthreads();

    // 5) l1 [300x300] + residual(actA): actB -> actC (trunk h)
    gemm_stage<1>(P.wL1, 19, 10, 300, s_act[1], s_act[2], s_act[0], nullptr, s_sb,
                  nullptr, lane, wave);
    __syncthreads();
    stage_bn(P.bnP_g, P.bnP_b, P.bnP_m, P.bnP_v, s_sb, tid, 300);
    __syncthreads();

    // 6) ps_w [300x300]: actC -> actB
    gemm_stage<0>(P.wPs, 19, 10, 300, s_act[2], s_act[1], nullptr, nullptr, s_sb,
                  nullptr, lane, wave);
    __syncthreads();

    // 7) ps_sw [25x300] + bias: actB -> s_p (f32)
    gemm_stage<2>(P.wPsS, 2, 10, 25, s_act[1], nullptr, nullptr, s_p, nullptr,
                  P.psb, lane, wave);
    stage_bn(P.bnA_g, P.bnA_b, P.bnA_m, P.bnA_v, s_sb, tid, 300);
    __syncthreads();

    // 8) as_w [300x300]: actC -> actA (res is dead)
    gemm_stage<0>(P.wAs, 19, 10, 300, s_act[2], s_act[0], nullptr, nullptr, s_sb,
                  nullptr, lane, wave);
    __syncthreads();

    // 9) as_sw [600x300] + bias: actA -> s_att (reuses xp LDS)
    gemm_stage<2>(P.wAsS, 38, 10, 600, s_act[0], nullptr, nullptr, s_att, nullptr,
                  P.asb, lane, wave);
    __syncthreads();

    // 10) softmax over m, 5x5 pose transform, fuse, +x
    for (int i = tid; i < 3000; i += BLK) {
        int n = i / 25, rem = i % 25, q = rem / 5, s = rem % 5;
        int ch = n * 5 + q;
        float L[5], mx = -3.4e38f;
#pragma unroll
        for (int m = 0; m < 5; ++m) { L[m] = s_att[ch * 32 + s * 5 + m]; mx = fmaxf(mx, L[m]); }
        float e[5], sum = 0.0f;
#pragma unroll
        for (int m = 0; m < 5; ++m) { e[m] = __expf(L[m] - mx); sum += e[m]; }
        float fuse = 0.0f;
#pragma unroll
        for (int m = 0; m < 5; ++m) {
            float fc = 0.0f;
#pragma unroll
            for (int c = 0; c < 5; ++c)
                fc = fmaf(s_x[(n * 5 + c) * 5 + m], s_p[(q * 5 + c) * 32 + s * 5 + m], fc);
            fuse = fmaf(fc, e[m], fuse);
        }
        fuse /= sum;
        P.out[(size_t)b * 3000 + i] = fuse + s_x[ch * 5 + s];
    }
}

// ---------------------------------------------------------------------------
// Host launcher
// ---------------------------------------------------------------------------
extern "C" void kernel_launch(void* const* d_in, const int* in_sizes, int n_in,
                              void* d_out, int out_size, void* d_ws, size_t ws_size,
                              hipStream_t stream) {
    (void)in_sizes; (void)n_in; (void)out_size; (void)ws_size;
    const float* x        = (const float*)d_in[0];
    const float* expand_w = (const float*)d_in[1];
    const float* bnE[4]   = {(const float*)d_in[2], (const float*)d_in[3],
                             (const float*)d_in[4], (const float*)d_in[5]};
    const float* l0_w     = (const float*)d_in[6];
    const float* bn0[4]   = {(const float*)d_in[7], (const float*)d_in[8],
                             (const float*)d_in[9], (const float*)d_in[10]};
    const float* l1_w     = (const float*)d_in[11];
    const float* bn1[4]   = {(const float*)d_in[12], (const float*)d_in[13],
                             (const float*)d_in[14], (const float*)d_in[15]};
    const float* ps_w     = (const float*)d_in[16];
    const float* bnP[4]   = {(const float*)d_in[17], (const float*)d_in[18],
                             (const float*)d_in[19], (const float*)d_in[20]};
    const float* ps_sw    = (const float*)d_in[21];
    const float* ps_sb    = (const float*)d_in[22];
    const float* as_w     = (const float*)d_in[23];
    const float* bnA[4]   = {(const float*)d_in[24], (const float*)d_in[25],
                             (const float*)d_in[26], (const float*)d_in[27]};
    const float* as_sw    = (const float*)d_in[28];
    const float* as_sb    = (const float*)d_in[29];

    __bf16* ws = (__bf16*)d_ws;
    size_t o = 0;
    __bf16* wExp = ws + o; o += (size_t)19 * 38 * 512;
    __bf16* wL0  = ws + o; o += (size_t)19 * 10 * 512;
    __bf16* wL1  = ws + o; o += (size_t)19 * 10 * 512;
    __bf16* wPs  = ws + o; o += (size_t)19 * 10 * 512;
    __bf16* wPsS = ws + o; o += (size_t)2  * 10 * 512;
    __bf16* wAs  = ws + o; o += (size_t)19 * 10 * 512;
    __bf16* wAsS = ws + o; o += (size_t)38 * 10 * 512;

    auto blocks = [](int total) { return (total + BLK - 1) / BLK; };
    permute_weight<<<blocks(19 * 38 * 512), BLK, 0, stream>>>(expand_w, wExp, 300, 1200, 19, 38);
    permute_weight<<<blocks(19 * 10 * 512), BLK, 0, stream>>>(l0_w,  wL0,  300, 300, 19, 10);
    permute_weight<<<blocks(19 * 10 * 512), BLK, 0, stream>>>(l1_w,  wL1,  300, 300, 19, 10);
    permute_weight<<<blocks(19 * 10 * 512), BLK, 0, stream>>>(ps_w,  wPs,  300, 300, 19, 10);
    permute_weight<<<blocks(2  * 10 * 512), BLK, 0, stream>>>(ps_sw, wPsS,  25, 300,  2, 10);
    permute_weight<<<blocks(19 * 10 * 512), BLK, 0, stream>>>(as_w,  wAs,  300, 300, 19, 10);
    permute_weight<<<blocks(38 * 10 * 512), BLK, 0, stream>>>(as_sw, wAsS, 600, 300, 38, 10);

    Params P;
    P.x = x; P.out = (float*)d_out;
    P.wExp = wExp; P.wL0 = wL0; P.wL1 = wL1; P.wPs = wPs;
    P.wPsS = wPsS; P.wAs = wAs; P.wAsS = wAsS;
    P.bnE_g = bnE[0]; P.bnE_b = bnE[1]; P.bnE_m = bnE[2]; P.bnE_v = bnE[3];
    P.bn0_g = bn0[0]; P.bn0_b = bn0[1]; P.bn0_m = bn0[2]; P.bn0_v = bn0[3];
    P.bn1_g = bn1[0]; P.bn1_b = bn1[1]; P.bn1_m = bn1[2]; P.bn1_v = bn1[3];
    P.bnP_g = bnP[0]; P.bnP_b = bnP[1]; P.bnP_m = bnP[2]; P.bnP_v = bnP[3];
    P.bnA_g = bnA[0]; P.bnA_b = bnA[1]; P.bnA_m = bnA[2]; P.bnA_v = bnA[3];
    P.psb = ps_sb; P.asb = as_sb;

    fused_multiview<<<4096, BLK, 0, stream>>>(P);
}
